// mamba_fusion3_26371099198483
// MI455X (gfx1250) — matrix operand order repaired and checked by reading, not compiled, not run on hardware
//
#include <hip/hip_runtime.h>
#include <hip/hip_bf16.h>
#include <math.h>

typedef __attribute__((ext_vector_type(16))) _Float16 v16h;
typedef __attribute__((ext_vector_type(8)))  _Float16 v8h;
typedef __attribute__((ext_vector_type(2)))  _Float16 v2h;
typedef __attribute__((ext_vector_type(8)))  float    v8f;

// ---- problem dims (fixed by reference) ----
#define B_SZ 2
#define C_SZ 768
#define D_SZ 768
#define L_SZ 4096            // H*W = 64*64
#define M_SZ 8192            // B*L
#define NIN  1536            // 2*D (in_proj output)
#define RNK  48              // dt_rank
#define NST  4               // d_state
#define NXP  56              // RNK + 2*NST
#define KCV  4               // conv kernel

// ---- chunked scan ----
#define NCH  16              // chunks along L
#define LCH  256             // L_SZ / NCH
#define BD_SZ (B_SZ * D_SZ)  // 1536 channels

// ---- WMMA GEMM tiling ----
#define TM 128
#define TN 64
#define TK 32
#define TKP (TK + 8)         // LDS row pad (bank skew)
#define TN2 128              // fast kernel N tile

// ---- workspace layout (per stream), 256B aligned bump allocator ----
constexpr size_t aln(size_t x) { return (x + 255) & ~size_t(255); }
constexpr size_t OFF_XN16  = 0;                                    // M*D f16 (aliased by g16 later)
constexpr size_t OFF_WIN16 = aln(OFF_XN16  + (size_t)M_SZ*D_SZ*2);
constexpr size_t OFF_WXP16 = aln(OFF_WIN16 + (size_t)D_SZ*NIN*2);
constexpr size_t OFF_WDT16 = aln(OFF_WXP16 + (size_t)D_SZ*NXP*2);
constexpr size_t OFF_WOUT16= aln(OFF_WDT16 + (size_t)RNK*D_SZ*2);
constexpr size_t OFF_UZ32  = aln(OFF_WOUT16+ (size_t)D_SZ*D_SZ*2); // M*1536 f32 (o32 aliases front)
constexpr size_t OFF_U16   = aln(OFF_UZ32  + (size_t)M_SZ*NIN*4);
constexpr size_t OFF_U32   = aln(OFF_U16   + (size_t)M_SZ*D_SZ*2);
constexpr size_t OFF_DBL32 = aln(OFF_U32   + (size_t)M_SZ*D_SZ*4);
constexpr size_t OFF_DTR16 = aln(OFF_DBL32 + (size_t)M_SZ*NXP*4);
constexpr size_t OFF_DT32  = aln(OFF_DTR16 + (size_t)M_SZ*RNK*2);
constexpr size_t STREAM_BYTES = aln(OFF_DT32 + (size_t)M_SZ*D_SZ*4);
// shared (stream-sequenced) scan summary buffers, after both stream regions
constexpr size_t SCAN_ELEMS = (size_t)NCH * BD_SZ * NST;           // 98304
constexpr size_t OFF_SCANP  = 2 * STREAM_BYTES;
constexpr size_t OFF_SCANH  = aln(OFF_SCANP + SCAN_ELEMS * 4);
constexpr size_t OFF_SCANI  = aln(OFF_SCANH + SCAN_ELEMS * 4);
// total ws use: ~264 MB + ~1.2 MB

// =====================================================================
// FAST GEMM: requires M%128==0, N%128==0, K%32==0 (in_proj, out_proj).
// 128x128 macro tile, 8 waves, each wave 32x64 (2x4 WMMA tiles).
// LDS double-buffered + register-pipelined staging: global_load_b128 of
// block k+1 issue BEFORE the 8 v_wmma of block k, and the matching LDS
// stores (and their s_wait_loadcnt) land AFTER them, so HBM/L2 latency
// hides behind matrix math.  No bounds checks -> EXEC all-ones at every
// v_wmma by construction.
// =====================================================================
__global__ __launch_bounds__(256)
void k_gemm_f16f32_fast(const _Float16* __restrict__ A, const _Float16* __restrict__ Bw,
                        float* __restrict__ C, int Mn, int Nn, int Kn) {
  __shared__ __align__(16) _Float16 As[2][TM][TKP];
  __shared__ __align__(16) _Float16 Bs[2][TN2][TKP];   // transposed: Bs[.][n][k]
  const int t    = threadIdx.x;
  const int lane = t & 31;
  const int wave = t >> 5;
  const int wm   = wave & 3;        // 4 waves in M (32 rows each)
  const int wn   = wave >> 2;       // 2 waves in N (64 cols each)
  const int lrow = lane & 15;
  const int hi   = lane >> 4;
  const int bm0  = blockIdx.y * TM;
  const int bn0  = blockIdx.x * TN2;

  // staging coordinates
  const int ar  = t >> 1;            // A: row 0..127
  const int aks = (t & 1) * 16;      // A: k segment {0,16}
  const int bkp = (t >> 4) * 2;      // B: k pair base 0,2,..,30
  const int bns = (t & 15) * 8;      // B: n segment 0..120

  v8f acc[2][4];
  #pragma unroll
  for (int i = 0; i < 2; ++i)
    #pragma unroll
    for (int j = 0; j < 4; ++j)
      #pragma unroll
      for (int e = 0; e < 8; ++e) acc[i][j][e] = 0.0f;

  // register staging (in flight between gload and lstore)
  v8h ra0, ra1, rb0, rb1;
  auto gload = [&](int k0) {
    const v8h* srcA = (const v8h*)(A + (size_t)(bm0 + ar) * Kn + k0 + aks);
    ra0 = srcA[0];
    ra1 = srcA[1];
    rb0 = *(const v8h*)(Bw + (size_t)(k0 + bkp)     * Nn + bn0 + bns);
    rb1 = *(const v8h*)(Bw + (size_t)(k0 + bkp + 1) * Nn + bn0 + bns);
  };
  auto lstore = [&](int buf) {
    _Float16* dst = &As[buf][ar][aks];
    *(v8h*)(dst)     = ra0;
    *(v8h*)(dst + 8) = ra1;
    #pragma unroll
    for (int i = 0; i < 8; ++i) {          // pairs of k -> ds_store_b32
      v2h pr; pr[0] = rb0[i]; pr[1] = rb1[i];
      *(v2h*)&Bs[buf][bns + i][bkp] = pr;  // 4B aligned (bkp even)
    }
  };

  const int kblocks = Kn / TK;
  gload(0);
  lstore(0);
  __syncthreads();
  for (int kb = 0; kb < kblocks; ++kb) {
    const int cur = kb & 1;
    const bool more = (kb + 1) < kblocks;
    if (more) gload((kb + 1) * TK);        // issue global loads, no wait yet

    v16h af[2], bf[4];
    #pragma unroll
    for (int mt = 0; mt < 2; ++mt) {
      const int row = wm * 32 + mt * 16 + lrow;
      const int ka  = hi * 8;
      v8h lo = *(const v8h*)&As[cur][row][ka];
      v8h hh = *(const v8h*)&As[cur][row][ka + 16];
      af[mt] = __builtin_shufflevector(lo, hh, 0,1,2,3,4,5,6,7,8,9,10,11,12,13,14,15);
    }
    #pragma unroll
    for (int nt = 0; nt < 4; ++nt) {
      const int col = wn * 64 + nt * 16 + lrow;
      const int kbb = hi * 8;
      v8h lo = *(const v8h*)&Bs[cur][col][kbb];
      v8h hh = *(const v8h*)&Bs[cur][col][kbb + 16];
      bf[nt] = __builtin_shufflevector(lo, hh, 0,1,2,3,4,5,6,7,8,9,10,11,12,13,14,15);
    }
    #pragma unroll
    for (int mt = 0; mt < 2; ++mt)
      #pragma unroll
      for (int nt = 0; nt < 4; ++nt)
        acc[mt][nt] = __builtin_amdgcn_wmma_f32_16x16x32_f16(
            false, af[mt], false, bf[nt], (short)0, acc[mt][nt], false, false);

    if (more) lstore(cur ^ 1);             // loadcnt wait + LDS stores after wmma
    __syncthreads();                       // publish nxt, retire reads of cur
  }
  #pragma unroll
  for (int mt = 0; mt < 2; ++mt) {
    #pragma unroll
    for (int nt = 0; nt < 4; ++nt) {
      const int gm0 = bm0 + wm * 32 + mt * 16 + hi * 8;
      const int gn  = bn0 + wn * 64 + nt * 16 + lrow;
      #pragma unroll
      for (int rr = 0; rr < 8; ++rr)
        C[(size_t)(gm0 + rr) * Nn + gn] = acc[mt][nt][rr];
    }
  }
}

// =====================================================================
// Generic bounds-checked GEMM (edge shapes: x_proj N=56, dt-proj K=48).
// =====================================================================
__global__ __launch_bounds__(256)
void k_gemm_f16f32(const _Float16* __restrict__ A, const _Float16* __restrict__ Bw,
                   float* __restrict__ C, int Mn, int Nn, int Kn) {
  __shared__ __align__(16) _Float16 As[TM][TKP];
  __shared__ __align__(16) _Float16 Bs[TN][TKP];   // transposed: Bs[n][k]
  const int t    = threadIdx.x;
  const int lane = t & 31;
  const int wave = t >> 5;
  const int wm   = wave & 3;
  const int wn   = wave >> 2;
  const int lrow = lane & 15;
  const int hi   = lane >> 4;
  const int bm0  = blockIdx.y * TM;
  const int bn0  = blockIdx.x * TN;

  v8f acc[2][2];
  #pragma unroll
  for (int i = 0; i < 2; ++i)
    #pragma unroll
    for (int j = 0; j < 2; ++j)
      #pragma unroll
      for (int e = 0; e < 8; ++e) acc[i][j][e] = 0.0f;

  const int ar  = t >> 1;
  const int aks = (t & 1) * 16;
  const int bkk = t >> 3;
  const int bnn = (t & 7) * 8;

  const int kblocks = (Kn + TK - 1) / TK;
  for (int kb = 0; kb < kblocks; ++kb) {
    const int k0 = kb * TK;
    {
      const int gm = bm0 + ar;
      const int gk = k0 + aks;
      _Float16* dst = &As[ar][aks];
      if (gm < Mn && (gk + 16) <= Kn) {
        const v8h* src = (const v8h*)(A + (size_t)gm * Kn + gk);  // K % 8 == 0 at all call sites
        *(v8h*)(dst)     = src[0];
        *(v8h*)(dst + 8) = src[1];
      } else {
        #pragma unroll
        for (int i = 0; i < 16; ++i) {
          const int kk = gk + i;
          dst[i] = (gm < Mn && kk < Kn) ? A[(size_t)gm * Kn + kk] : (_Float16)0.0f;
        }
      }
    }
    {
      const int gk = k0 + bkk;
      if ((bn0 + TN) <= Nn && (k0 + TK) <= Kn) {     // block-uniform -> scalar branch
        v8h v = *(const v8h*)(Bw + (size_t)gk * Nn + bn0 + bnn);
        #pragma unroll
        for (int i = 0; i < 8; ++i) Bs[bnn + i][bkk] = v[i];
      } else {
        #pragma unroll
        for (int i = 0; i < 8; ++i) {
          const int gn = bn0 + bnn + i;
          Bs[bnn + i][bkk] = (gk < Kn && gn < Nn) ? Bw[(size_t)gk * Nn + gn] : (_Float16)0.0f;
        }
      }
    }
    __syncthreads();
    v16h af[2], bf[2];
    #pragma unroll
    for (int mt = 0; mt < 2; ++mt) {
      const int row = wm * 32 + mt * 16 + lrow;
      const int ka  = hi * 8;
      v8h lo = *(const v8h*)&As[row][ka];
      v8h hh = *(const v8h*)&As[row][ka + 16];
      af[mt] = __builtin_shufflevector(lo, hh, 0,1,2,3,4,5,6,7,8,9,10,11,12,13,14,15);
    }
    #pragma unroll
    for (int nt = 0; nt < 2; ++nt) {
      const int col = wn * 32 + nt * 16 + lrow;
      const int kbb = hi * 8;
      v8h lo = *(const v8h*)&Bs[col][kbb];
      v8h hh = *(const v8h*)&Bs[col][kbb + 16];
      bf[nt] = __builtin_shufflevector(lo, hh, 0,1,2,3,4,5,6,7,8,9,10,11,12,13,14,15);
    }
    #pragma unroll
    for (int mt = 0; mt < 2; ++mt)
      #pragma unroll
      for (int nt = 0; nt < 2; ++nt)
        acc[mt][nt] = __builtin_amdgcn_wmma_f32_16x16x32_f16(
            false, af[mt], false, bf[nt], (short)0, acc[mt][nt], false, false);
    __syncthreads();
  }
  #pragma unroll
  for (int mt = 0; mt < 2; ++mt) {
    #pragma unroll
    for (int nt = 0; nt < 2; ++nt) {
      const int gm0 = bm0 + wm * 32 + mt * 16 + hi * 8;
      const int gn  = bn0 + wn * 32 + nt * 16 + lrow;
      if (gn < Nn) {
        #pragma unroll
        for (int rr = 0; rr < 8; ++rr) {
          const int gm = gm0 + rr;
          if (gm < Mn) C[(size_t)gm * Nn + gn] = acc[mt][nt][rr];
        }
      }
    }
  }
}

// ---- f32 -> f16 convert (weights) ----
__global__ void k_f32_to_f16(const float* __restrict__ in, _Float16* __restrict__ out, int n) {
  const int i = blockIdx.x * 256 + threadIdx.x;
  if (i < n) out[i] = (_Float16)in[i];
}

// ---- LayerNorm over C with BCHW->row gather, emit f16 rows ----
__global__ __launch_bounds__(256)
void k_layernorm(const float* __restrict__ x, const float* __restrict__ g,
                 const float* __restrict__ be, _Float16* __restrict__ xn) {
  __shared__ float s1[256];
  __shared__ float s2[256];
  const int row = blockIdx.x;            // b*L + l
  const int b = row / L_SZ;
  const int l = row - b * L_SZ;
  const int t = threadIdx.x;
  float v0[3];
  float sum = 0.f, sq = 0.f;
  #pragma unroll
  for (int i = 0; i < 3; ++i) {
    const int c = t + i * 256;
    const float v = x[((size_t)b * C_SZ + c) * L_SZ + l];
    v0[i] = v; sum += v; sq += v * v;
  }
  s1[t] = sum; s2[t] = sq;
  __syncthreads();
  for (int off = 128; off > 0; off >>= 1) {
    if (t < off) { s1[t] += s1[t + off]; s2[t] += s2[t + off]; }
    __syncthreads();
  }
  const float mean = s1[0] * (1.0f / C_SZ);
  const float var  = s2[0] * (1.0f / C_SZ) - mean * mean;
  const float rs   = rsqrtf(var + 1e-5f);
  #pragma unroll
  for (int i = 0; i < 3; ++i) {
    const int c = t + i * 256;
    xn[(size_t)row * C_SZ + c] = (_Float16)((v0[i] - mean) * rs * g[c] + be[c]);
  }
}

// ---- depthwise causal conv (K=4) + SiLU; emit f16 (GEMM A) and f32 (scan) ----
__global__ __launch_bounds__(256)
void k_conv_silu(const float* __restrict__ uz, const float* __restrict__ wc,
                 const float* __restrict__ bc, _Float16* __restrict__ u16,
                 float* __restrict__ u32) {
  const int idx = blockIdx.x * 256 + threadIdx.x;   // M*D
  const int row = idx / D_SZ;
  const int d   = idx - row * D_SZ;
  const int b   = row / L_SZ;
  const int l   = row - b * L_SZ;
  float acc = bc[d];
  #pragma unroll
  for (int j = 0; j < KCV; ++j) {
    const int ls = l + j - (KCV - 1);
    if (ls >= 0)
      acc += wc[d * KCV + j] * uz[((size_t)b * L_SZ + ls) * NIN + d];
  }
  const float s = acc / (1.0f + __expf(-acc));
  u16[(size_t)row * D_SZ + d] = (_Float16)s;
  u32[(size_t)row * D_SZ + d] = s;
}

// ---- slice dtr (first 48 cols of dbl) to f16 for the dt GEMM ----
__global__ void k_extract_dtr(const float* __restrict__ dbl, _Float16* __restrict__ dtr) {
  const int idx = blockIdx.x * 256 + threadIdx.x;   // M*RNK
  const int r = idx / RNK;
  const int c = idx - r * RNK;
  dtr[idx] = (_Float16)dbl[(size_t)r * NXP + c];
}

// ---- dt = softplus(dt + bdt), in place ----
__global__ void k_dt_softplus(float* __restrict__ dt, const float* __restrict__ bdt) {
  const int idx = blockIdx.x * 256 + threadIdx.x;   // M*D
  const int d = idx % D_SZ;
  const float v = dt[idx] + bdt[d];
  dt[idx] = (v > 20.0f) ? v : log1pf(__expf(v));
}

// =====================================================================
// Chunked selective scan (3 passes): per-chunk (P,H) summaries, serial
// combine over 16 chunk summaries, replay with correct incoming state.
// =====================================================================
__global__ __launch_bounds__(256)
void k_scan_chunk(const float* __restrict__ dt, const float* __restrict__ u,
                  const float* __restrict__ dblS, const float* __restrict__ Alog,
                  float* __restrict__ P, float* __restrict__ Hc) {
  const int bd = blockIdx.x * 256 + threadIdx.x;   // 0..BD_SZ-1 (coalesced in d)
  const int ch = blockIdx.y;
  const int b = bd / D_SZ;
  const int d = bd - b * D_SZ;
  float An[NST];
  #pragma unroll
  for (int n = 0; n < NST; ++n) An[n] = -__expf(Alog[d * NST + n]);
  float h[NST] = {0.f, 0.f, 0.f, 0.f};
  float p[NST] = {1.f, 1.f, 1.f, 1.f};
  const int l0 = ch * LCH;
  for (int l = l0; l < l0 + LCH; ++l) {
    const size_t row = (size_t)b * L_SZ + l;
    const float dtv = dt[row * D_SZ + d];
    const float dtu = dtv * u[row * D_SZ + d];
    #pragma unroll
    for (int n = 0; n < NST; ++n) {
      const float dA = __expf(dtv * An[n]);
      const float Bn = dblS[row * NXP + RNK + n];
      p[n] *= dA;
      h[n] = dA * h[n] + dtu * Bn;
    }
  }
  const size_t o = ((size_t)ch * BD_SZ + bd) * NST;
  #pragma unroll
  for (int n = 0; n < NST; ++n) { P[o + n] = p[n]; Hc[o + n] = h[n]; }
}

__global__ __launch_bounds__(256)
void k_scan_combine(const float* __restrict__ P, const float* __restrict__ Hc,
                    float* __restrict__ Hin) {
  const int bd = blockIdx.x * 256 + threadIdx.x;
  float hc[NST] = {0.f, 0.f, 0.f, 0.f};
  for (int ch = 0; ch < NCH; ++ch) {
    const size_t o = ((size_t)ch * BD_SZ + bd) * NST;
    #pragma unroll
    for (int n = 0; n < NST; ++n) {
      Hin[o + n] = hc[n];
      hc[n] = P[o + n] * hc[n] + Hc[o + n];
    }
  }
}

__global__ __launch_bounds__(256)
void k_scan_apply(const float* __restrict__ dt, const float* __restrict__ u,
                  const float* __restrict__ uz, const float* __restrict__ dblS,
                  const float* __restrict__ dblO, const float* __restrict__ Alog,
                  const float* __restrict__ Dskip, const float* __restrict__ Hin,
                  _Float16* __restrict__ g16) {
  const int bd = blockIdx.x * 256 + threadIdx.x;
  const int ch = blockIdx.y;
  const int b = bd / D_SZ;
  const int d = bd - b * D_SZ;
  float An[NST];
  #pragma unroll
  for (int n = 0; n < NST; ++n) An[n] = -__expf(Alog[d * NST + n]);
  float h[NST];
  {
    const size_t o = ((size_t)ch * BD_SZ + bd) * NST;
    #pragma unroll
    for (int n = 0; n < NST; ++n) h[n] = Hin[o + n];
  }
  const float dsk = Dskip[d];
  const int l0 = ch * LCH;
  for (int l = l0; l < l0 + LCH; ++l) {
    const size_t row = (size_t)b * L_SZ + l;
    const float dtv = dt[row * D_SZ + d];
    const float uv  = u[row * D_SZ + d];
    const float dtu = dtv * uv;
    float y = 0.f;
    #pragma unroll
    for (int n = 0; n < NST; ++n) {
      const float dA = __expf(dtv * An[n]);
      const float Bn = dblS[row * NXP + RNK + n];
      const float Cn = dblO[row * NXP + RNK + NST + n];   // other stream's C
      h[n] = dA * h[n] + dtu * Bn;
      y += h[n] * Cn;
    }
    y += uv * dsk;
    const float z  = uz[row * NIN + D_SZ + d];
    const float sz = z / (1.0f + __expf(-z));
    g16[row * D_SZ + d] = (_Float16)(y * sz);
  }
}

// ---- residual: out[b,l,c] = x[b,c,l] + o[b,l,c] ----
__global__ __launch_bounds__(256)
void k_residual(const float* __restrict__ x, const float* __restrict__ o,
                float* __restrict__ out) {
  const int idx = blockIdx.x * 256 + threadIdx.x;   // M*C
  const int row = idx / C_SZ;
  const int c   = idx - row * C_SZ;
  const int b   = row / L_SZ;
  const int l   = row - b * L_SZ;
  out[idx] = x[((size_t)b * C_SZ + c) * L_SZ + l] + o[idx];
}

extern "C" void kernel_launch(void* const* d_in, const int* in_sizes, int n_in,
                              void* d_out, int out_size, void* d_ws, size_t ws_size,
                              hipStream_t stream) {
  (void)in_sizes; (void)n_in; (void)out_size; (void)ws_size;
  char* ws = (char*)d_ws;

  const float* xin[2]   = {(const float*)d_in[0], (const float*)d_in[1]};
  const float *ln_g[2], *ln_b[2], *Win[2], *Wconv[2], *bconv[2], *Wxp[2],
              *Wdt[2], *bdt[2], *Alog[2], *Dsk[2], *Wout[2];
  _Float16 *xn16[2], *win16[2], *wxp16[2], *wdt16[2], *wout16[2], *u16[2], *dtr16[2], *g16[2];
  float *uz32[2], *u32[2], *dbl32[2], *dt32[2], *o32[2];

  for (int s = 0; s < 2; ++s) {
    const int i0 = 2 + s * 11;
    ln_g[s]  = (const float*)d_in[i0 + 0];
    ln_b[s]  = (const float*)d_in[i0 + 1];
    Win[s]   = (const float*)d_in[i0 + 2];
    Wconv[s] = (const float*)d_in[i0 + 3];
    bconv[s] = (const float*)d_in[i0 + 4];
    Wxp[s]   = (const float*)d_in[i0 + 5];
    Wdt[s]   = (const float*)d_in[i0 + 6];
    bdt[s]   = (const float*)d_in[i0 + 7];
    Alog[s]  = (const float*)d_in[i0 + 8];
    Dsk[s]   = (const float*)d_in[i0 + 9];
    Wout[s]  = (const float*)d_in[i0 + 10];
    char* w = ws + (size_t)s * STREAM_BYTES;
    xn16[s]   = (_Float16*)(w + OFF_XN16);
    win16[s]  = (_Float16*)(w + OFF_WIN16);
    wxp16[s]  = (_Float16*)(w + OFF_WXP16);
    wdt16[s]  = (_Float16*)(w + OFF_WDT16);
    wout16[s] = (_Float16*)(w + OFF_WOUT16);
    uz32[s]   = (float*)(w + OFF_UZ32);
    u16[s]    = (_Float16*)(w + OFF_U16);
    u32[s]    = (float*)(w + OFF_U32);
    dbl32[s]  = (float*)(w + OFF_DBL32);
    dtr16[s]  = (_Float16*)(w + OFF_DTR16);
    dt32[s]   = (float*)(w + OFF_DT32);
    g16[s]    = xn16[s];   // alias: xn dead after in_proj GEMM, scan writes g later
    o32[s]    = uz32[s];   // alias: uz dead after scan, out GEMM writes o later
  }
  float* scanP = (float*)(ws + OFF_SCANP);   // shared, streams sequenced
  float* scanH = (float*)(ws + OFF_SCANH);
  float* scanI = (float*)(ws + OFF_SCANI);

  auto cvt = [&](const float* src, _Float16* dst, int n) {
    k_f32_to_f16<<<(n + 255) / 256, 256, 0, stream>>>(src, dst, n);
  };

  // ---- phase 1: per-stream pre-processing up to dt ----
  for (int s = 0; s < 2; ++s) {
    cvt(Win[s],  win16[s],  D_SZ * NIN);
    cvt(Wxp[s],  wxp16[s],  D_SZ * NXP);
    cvt(Wdt[s],  wdt16[s],  RNK * D_SZ);
    cvt(Wout[s], wout16[s], D_SZ * D_SZ);

    k_layernorm<<<M_SZ, 256, 0, stream>>>(xin[s], ln_g[s], ln_b[s], xn16[s]);

    // in_proj: (8192x768) @ (768x1536) -> uz   [aligned fast path]
    k_gemm_f16f32_fast<<<dim3(NIN / TN2, M_SZ / TM), 256, 0, stream>>>(
        xn16[s], win16[s], uz32[s], M_SZ, NIN, D_SZ);

    k_conv_silu<<<(M_SZ * D_SZ) / 256, 256, 0, stream>>>(
        uz32[s], Wconv[s], bconv[s], u16[s], u32[s]);

    // x_proj: (8192x768) @ (768x56) -> dbl   [generic, N edge]
    k_gemm_f16f32<<<dim3((NXP + TN - 1) / TN, M_SZ / TM), 256, 0, stream>>>(
        u16[s], wxp16[s], dbl32[s], M_SZ, NXP, D_SZ);

    k_extract_dtr<<<(M_SZ * RNK) / 256, 256, 0, stream>>>(dbl32[s], dtr16[s]);

    // dt proj: (8192x48) @ (48x768) -> dt   [generic, K edge]
    k_gemm_f16f32<<<dim3(D_SZ / TN, M_SZ / TM), 256, 0, stream>>>(
        dtr16[s], wdt16[s], dt32[s], M_SZ, D_SZ, RNK);

    k_dt_softplus<<<(M_SZ * D_SZ) / 256, 256, 0, stream>>>(dt32[s], bdt[s]);
  }

  // ---- phase 2: chunked cross selective scans (stream s reads other's C) ----
  for (int s = 0; s < 2; ++s) {
    k_scan_chunk<<<dim3(BD_SZ / 256, NCH), 256, 0, stream>>>(
        dt32[s], u32[s], dbl32[s], Alog[s], scanP, scanH);
    k_scan_combine<<<BD_SZ / 256, 256, 0, stream>>>(scanP, scanH, scanI);
    k_scan_apply<<<dim3(BD_SZ / 256, NCH), 256, 0, stream>>>(
        dt32[s], u32[s], uz32[s], dbl32[s], dbl32[1 - s], Alog[s], Dsk[s],
        scanI, g16[s]);
  }

  // ---- phase 3: out_proj GEMM + residual ----
  for (int s = 0; s < 2; ++s) {
    // out_proj: (8192x768) @ (768x768) -> o   [aligned fast path]
    k_gemm_f16f32_fast<<<dim3(D_SZ / TN2, M_SZ / TM), 256, 0, stream>>>(
        g16[s], wout16[s], o32[s], M_SZ, D_SZ, D_SZ);
    float* outS = (float*)d_out + (size_t)s * M_SZ * C_SZ;
    k_residual<<<(M_SZ * C_SZ) / 256, 256, 0, stream>>>(xin[s], o32[s], outS);
  }
}
